// GraphormerAttention_7842610282921
// MI455X (gfx1250) — matrix operand order repaired and checked by reading, not compiled, not run on hardware
//
#include <hip/hip_runtime.h>

// ---------------------------------------------------------------------------
// Graphormer attention for gfx1250 (MI455X), bf16 WMMA pipeline.
// B=8, S=1024, H=768, heads=12, hd=64.
// ---------------------------------------------------------------------------

typedef __attribute__((ext_vector_type(16))) __bf16 v16bf;
typedef __attribute__((ext_vector_type(8)))  __bf16 v8bf;
typedef __attribute__((ext_vector_type(8)))  float  v8f;

// Types for the async global->LDS builtin (from the compiler diagnostic:
// param 0 is a vector-of-4-int pointer in the global address space).
typedef __attribute__((__vector_size__(4 * sizeof(int)))) int v4i;
typedef v4i __attribute__((address_space(1)))* gas_v4i_ptr;
typedef v4i __attribute__((address_space(3)))* las_v4i_ptr;

#define HIDDEN 768
#define NHEADS 12
#define HDIM   64
#define BATCH  8
#define SEQ    1024
#define NTOK   (BATCH * SEQ)   // 8192

// Workspace layout (units: __bf16 elements)
static constexpr size_t OFF_XB = 0;                                  // [8192][768]
static constexpr size_t OFF_WQ = OFF_XB + (size_t)NTOK * HIDDEN;     // [768][768]
static constexpr size_t OFF_WK = OFF_WQ + (size_t)HIDDEN * HIDDEN;
static constexpr size_t OFF_WV = OFF_WK + (size_t)HIDDEN * HIDDEN;
static constexpr size_t OFF_WO = OFF_WV + (size_t)HIDDEN * HIDDEN;
static constexpr size_t OFF_Q  = OFF_WO + (size_t)HIDDEN * HIDDEN;   // [B,nh,S,hd]
static constexpr size_t OFF_K  = OFF_Q  + (size_t)NTOK * HIDDEN;     // [B,nh,S,hd]
static constexpr size_t OFF_VT = OFF_K  + (size_t)NTOK * HIDDEN;     // [B,nh,hd,S]
static constexpr size_t OFF_Y  = OFF_VT + (size_t)NTOK * HIDDEN;     // [B,S,H]

// ---- WMMA helpers ---------------------------------------------------------

__device__ __forceinline__ v8f wmma_bf16(v16bf a, v16bf b, v8f c) {
  // D = A(16x32) * B(32x16) + C, fp32 accumulate
  return __builtin_amdgcn_wmma_f32_16x16x32_bf16(
      /*neg_a=*/false, a, /*neg_b=*/false, b,
      /*c_mod=*/(short)0, c, /*reuse_a=*/false, /*reuse_b=*/false);
}

// A fragment: 16x32 tile, row-major source [rows][ld], base at (row0, k0).
// a[8h+t] = A[m][16h + 8*kh + t], m = lane&15, kh = lane>>4  (ISA 7.12.2)
__device__ __forceinline__ v16bf load_afrag(const __bf16* base, int ld, int lane) {
  const int m  = lane & 15;
  const int kh = lane >> 4;
  const __bf16* p = base + (size_t)m * ld + 8 * kh;
  v16bf a;
  ((v8bf*)&a)[0] = *(const v8bf*)(p);
  ((v8bf*)&a)[1] = *(const v8bf*)(p + 16);
  return a;
}

// B fragment for B[k][n] given row-major [N][K] storage (i.e. B = Wrows^T):
// b[8h+t] = W[n0 + (lane&15)][k0 + 16*h + 8*(lane>>4) + t]  (matches A-frag k order).
__device__ __forceinline__ v16bf load_bfrag(const __bf16* base, int ld, int lane) {
  const int nn = lane & 15;
  const int kg = lane >> 4;
  const __bf16* p = base + (size_t)nn * ld + 8 * kg;
  v16bf b;
  ((v8bf*)&b)[0] = *(const v8bf*)(p);
  ((v8bf*)&b)[1] = *(const v8bf*)(p + 16);
  return b;
}

// Cooperative stage of one 16x768 bf16 A-tile (24576 B) into LDS by 128
// threads. Uses the CDNA5 async global->LDS path when available (ASYNCcnt),
// else a plain vector copy.
__device__ __forceinline__ void stage_atile(const __bf16* __restrict__ gsrc,
                                            __bf16* lds, int tid) {
#if __has_builtin(__builtin_amdgcn_global_load_async_to_lds_b128)
#pragma unroll
  for (int i = 0; i < 12; ++i) {
    const int off = (tid + i * 128) * 16;   // bytes
    __builtin_amdgcn_global_load_async_to_lds_b128(
        (gas_v4i_ptr)((char*)gsrc + off),
        (las_v4i_ptr)((char*)lds + off),
        /*offset=*/0, /*cpol=*/0);
  }
#if __has_builtin(__builtin_amdgcn_s_wait_asynccnt)
  __builtin_amdgcn_s_wait_asynccnt(0);
#else
  asm volatile("s_wait_asynccnt 0x0" ::: "memory");
#endif
#else
#pragma unroll
  for (int i = 0; i < 12; ++i) {
    const int off = (tid + i * 128) * 8;    // elements (16 B chunks)
    *(v8bf*)(lds + off) = *(const v8bf*)(gsrc + off);
  }
#endif
  __syncthreads();
}

// ---- Kernel 0: fp32 -> bf16 conversion of x and weights -------------------

__global__ void cvt_kernel(const float* __restrict__ x,
                           const float* __restrict__ wq, const float* __restrict__ wk,
                           const float* __restrict__ wv, const float* __restrict__ wo,
                           __bf16* __restrict__ ws) {
  size_t i = (size_t)blockIdx.x * blockDim.x + threadIdx.x;
  size_t stride = (size_t)gridDim.x * blockDim.x;
  for (size_t t = i; t < (size_t)NTOK * HIDDEN; t += stride)
    ws[OFF_XB + t] = (__bf16)x[t];
  for (size_t t = i; t < (size_t)HIDDEN * HIDDEN; t += stride) {
    ws[OFF_WQ + t] = (__bf16)wq[t];
    ws[OFF_WK + t] = (__bf16)wk[t];
    ws[OFF_WV + t] = (__bf16)wv[t];
    ws[OFF_WO + t] = (__bf16)wo[t];
  }
}

// ---- Kernel 1: QKV projection (x @ W^T + b), wave = 16x64 tile ------------
// All 4 waves of a block share (which, mtile); A-tile staged once in LDS.

__global__ void qkv_kernel(const float* __restrict__ bq, const float* __restrict__ bk,
                           const float* __restrict__ bv, __bf16* __restrict__ ws) {
  __shared__ __align__(16) __bf16 atile[16 * HIDDEN];   // 24 KB

  const int lane  = threadIdx.x & 31;
  const int wid   = blockIdx.x * (blockDim.x >> 5) + (threadIdx.x >> 5);
  // 3 weights x 512 m-tiles x 12 n-groups(64)
  const int which = wid / (512 * 12);
  const int rem   = wid % (512 * 12);
  const int mtile = rem / 12;
  const int ngrp  = rem % 12;

  const __bf16* Ag = ws + OFF_XB + (size_t)mtile * 16 * HIDDEN;
  stage_atile(Ag, atile, threadIdx.x);

  const size_t woff = (which == 0) ? OFF_WQ : (which == 1) ? OFF_WK : OFF_WV;
  const __bf16* W = ws + woff + (size_t)ngrp * 64 * HIDDEN;

  const int nn = lane & 15;
  const int hf = lane >> 4;

  v8f zero = {};
  v8f acc[4];
  acc[0] = zero; acc[1] = zero; acc[2] = zero; acc[3] = zero;

  for (int k0 = 0; k0 < HIDDEN; k0 += 32) {
    if (k0 + 64 < HIDDEN)
      __builtin_prefetch(W + (size_t)nn * HIDDEN + k0 + 64, 0, 1);
    v16bf a = load_afrag(atile + k0, HIDDEN, lane);
#pragma unroll
    for (int j = 0; j < 4; ++j) {
      v16bf b = load_bfrag(W + (size_t)(16 * j) * HIDDEN + k0, HIDDEN, lane);
      acc[j] = wmma_bf16(a, b, acc[j]);
    }
  }

  const float* bias = (which == 0) ? bq : (which == 1) ? bk : bv;

  if (which < 2) {
    __bf16* dst = ws + ((which == 0) ? OFF_Q : OFF_K);
#pragma unroll
    for (int j = 0; j < 4; ++j) {
      float bj = bias[ngrp * 64 + j * 16 + nn];
#pragma unroll
      for (int r = 0; r < 8; ++r) {
        int mg = mtile * 16 + r + 8 * hf;           // global token
        int bb = mg >> 10, ss = mg & 1023;
        dst[(((size_t)bb * NHEADS + ngrp) * SEQ + ss) * HDIM + j * 16 + nn] =
            (__bf16)(acc[j][r] + bj);
      }
    }
  } else {
    // V stored transposed: [B, nh, hd, S]; per lane 8 contiguous tokens.
    __bf16* dst = ws + OFF_VT;
    const int bb = (mtile * 16) >> 10;
    const int s0 = (mtile * 16) & 1023;
#pragma unroll
    for (int j = 0; j < 4; ++j) {
      float bj = bias[ngrp * 64 + j * 16 + nn];
      v8bf pk;
#pragma unroll
      for (int r = 0; r < 8; ++r) pk[r] = (__bf16)(acc[j][r] + bj);
      *(v8bf*)&dst[(((size_t)bb * NHEADS + ngrp) * HDIM + j * 16 + nn) * SEQ +
                   s0 + 8 * hf] = pk;
    }
  }
}

// ---- Kernel 2: flash attention, wave = (b, h, 16-query tile) --------------

__global__ void attn_kernel(const float* __restrict__ sp, const float* __restrict__ ee,
                            const unsigned char* __restrict__ mask,
                            __bf16* __restrict__ ws) {
  __shared__ __align__(16) __bf16 pbuf[4][16 * 32];   // per-wave P transpose buffer

  const int wslot = threadIdx.x >> 5;
  const int lane  = threadIdx.x & 31;
  const int wid   = blockIdx.x * (blockDim.x >> 5) + wslot;
  const int qt = wid & 63;
  const int bh = wid >> 6;
  const int bb = bh / NHEADS;
  const int hh = bh % NHEADS;
  const int nn = lane & 15;
  const int hf = lane >> 4;
  const int q0 = qt * 16;

  const __bf16* Qb = ws + OFF_Q  + ((size_t)bh * SEQ + q0) * HDIM;
  const __bf16* Kb = ws + OFF_K  + (size_t)bh * SEQ * HDIM;
  const __bf16* Vt = ws + OFF_VT + (size_t)bh * HDIM * SEQ;
  const float* spb = sp + (size_t)bb * SEQ * SEQ;
  const float* eeb = ee + (size_t)bb * SEQ * SEQ;
  const unsigned char* mb = mask + (size_t)bb * SEQ * SEQ;

  const v16bf qa0 = load_afrag(Qb, HDIM, lane);        // d = 0..31
  const v16bf qa1 = load_afrag(Qb + 32, HDIM, lane);   // d = 32..63

  v8f zero = {};
  v8f o[4];
  o[0] = zero; o[1] = zero; o[2] = zero; o[3] = zero;
  float mrow[8], lrow[8];
#pragma unroll
  for (int r = 0; r < 8; ++r) { mrow[r] = -3.0e38f; lrow[r] = 0.0f; }

  __bf16* pb = &pbuf[wslot][0];

  for (int t2 = 0; t2 < SEQ / 32; ++t2) {
    const int k0 = t2 * 32;
    // ---- scores: two 16x16 tiles (32 keys) ----
    v8f s0 = zero, s1 = zero;
    {
      const __bf16* Kt0 = Kb + (size_t)k0 * HDIM;
      const __bf16* Kt1 = Kb + (size_t)(k0 + 16) * HDIM;
      if (k0 + 32 < SEQ)
        __builtin_prefetch(Kb + (size_t)(k0 + 32) * HDIM + (size_t)nn * HDIM, 0, 1);
      s0 = wmma_bf16(qa0, load_bfrag(Kt0, HDIM, lane), s0);
      s0 = wmma_bf16(qa1, load_bfrag(Kt0 + 32, HDIM, lane), s0);
      s1 = wmma_bf16(qa0, load_bfrag(Kt1, HDIM, lane), s1);
      s1 = wmma_bf16(qa1, load_bfrag(Kt1 + 32, HDIM, lane), s1);
    }
    // ---- scale + graph biases + mask ----
#pragma unroll
    for (int r = 0; r < 8; ++r) {
      const int qrow = q0 + r + 8 * hf;
      const size_t ro = (size_t)qrow * SEQ;
      const int kc0 = k0 + nn, kc1 = k0 + 16 + nn;
      float v0 = s0[r] * 0.125f + spb[ro + kc0] + eeb[ro + kc0];
      float v1 = s1[r] * 0.125f + spb[ro + kc1] + eeb[ro + kc1];
      if (mb[ro + kc0]) v0 = -3.0e38f;
      if (mb[ro + kc1]) v1 = -3.0e38f;
      s0[r] = v0; s1[r] = v1;
    }
    // ---- online softmax: row max (reduce within 16-lane halves) ----
    float tmax[8];
#pragma unroll
    for (int r = 0; r < 8; ++r) tmax[r] = fmaxf(s0[r], s1[r]);
#pragma unroll
    for (int xm = 1; xm < 16; xm <<= 1)
#pragma unroll
      for (int r = 0; r < 8; ++r)
        tmax[r] = fmaxf(tmax[r], __shfl_xor(tmax[r], xm, 32));

    float tsum[8];
#pragma unroll
    for (int r = 0; r < 8; ++r) {
      const float mn = fmaxf(mrow[r], tmax[r]);
      const float sc = __expf(mrow[r] - mn);
      const float p0 = __expf(s0[r] - mn);
      const float p1 = __expf(s1[r] - mn);
      lrow[r] *= sc;
      o[0][r] *= sc; o[1][r] *= sc; o[2][r] *= sc; o[3][r] *= sc;
      mrow[r] = mn;
      tsum[r] = p0 + p1;
      const int prow = r + 8 * hf;                   // matches C-layout row
      pb[prow * 32 + nn]      = (__bf16)p0;
      pb[prow * 32 + 16 + nn] = (__bf16)p1;
    }
#pragma unroll
    for (int xm = 1; xm < 16; xm <<= 1)
#pragma unroll
      for (int r = 0; r < 8; ++r) tsum[r] += __shfl_xor(tsum[r], xm, 32);
#pragma unroll
    for (int r = 0; r < 8; ++r) lrow[r] += tsum[r];

    // ---- P(16x32) @ V(32x64): A-frag from LDS, B-frags from Vt ----
    const v16bf pa = load_afrag(pb, 32, lane);
#pragma unroll
    for (int j = 0; j < 4; ++j) {
      v16bf vb = load_bfrag(Vt + (size_t)(16 * j) * SEQ + k0, SEQ, lane);
      o[j] = wmma_bf16(pa, vb, o[j]);
    }
  }

  // ---- normalize and store Y[b, s, h*64+d] as bf16 ----
  __bf16* Y = ws + OFF_Y;
#pragma unroll
  for (int j = 0; j < 4; ++j)
#pragma unroll
    for (int r = 0; r < 8; ++r) {
      const int ss = q0 + r + 8 * hf;
      const float val = o[j][r] / lrow[r];
      Y[((size_t)bb * SEQ + ss) * HIDDEN + hh * HDIM + j * 16 + nn] = (__bf16)val;
    }
}

// ---- Kernel 3: output projection (Y @ Wo^T + bo), fp32 out ----------------

__global__ void out_kernel(const float* __restrict__ bo,
                           const __bf16* __restrict__ ws,
                           float* __restrict__ out) {
  __shared__ __align__(16) __bf16 atile[16 * HIDDEN];   // 24 KB

  const int lane  = threadIdx.x & 31;
  const int wid   = blockIdx.x * (blockDim.x >> 5) + (threadIdx.x >> 5);
  const int mtile = wid / 12;
  const int ngrp  = wid % 12;
  const int nn = lane & 15;
  const int hf = lane >> 4;

  const __bf16* Ag = ws + OFF_Y + (size_t)mtile * 16 * HIDDEN;
  stage_atile(Ag, atile, threadIdx.x);

  const __bf16* W = ws + OFF_WO + (size_t)ngrp * 64 * HIDDEN;

  v8f zero = {};
  v8f acc[4];
  acc[0] = zero; acc[1] = zero; acc[2] = zero; acc[3] = zero;

  for (int k0 = 0; k0 < HIDDEN; k0 += 32) {
    if (k0 + 64 < HIDDEN)
      __builtin_prefetch(W + (size_t)nn * HIDDEN + k0 + 64, 0, 1);
    v16bf a = load_afrag(atile + k0, HIDDEN, lane);
#pragma unroll
    for (int j = 0; j < 4; ++j) {
      v16bf b = load_bfrag(W + (size_t)(16 * j) * HIDDEN + k0, HIDDEN, lane);
      acc[j] = wmma_bf16(a, b, acc[j]);
    }
  }

#pragma unroll
  for (int j = 0; j < 4; ++j) {
    const float bj = bo[ngrp * 64 + j * 16 + nn];
#pragma unroll
    for (int r = 0; r < 8; ++r) {
      const int mg = mtile * 16 + r + 8 * hf;
      out[(size_t)mg * HIDDEN + ngrp * 64 + j * 16 + nn] = acc[j][r] + bj;
    }
  }
}

// ---------------------------------------------------------------------------

extern "C" void kernel_launch(void* const* d_in, const int* in_sizes, int n_in,
                              void* d_out, int out_size, void* d_ws, size_t ws_size,
                              hipStream_t stream) {
  (void)in_sizes; (void)n_in; (void)out_size; (void)ws_size;
  const float* x  = (const float*)d_in[0];
  const float* sp = (const float*)d_in[1];
  const float* ee = (const float*)d_in[2];
  const unsigned char* mask = (const unsigned char*)d_in[3];
  const float* Wq = (const float*)d_in[4];
  const float* bq = (const float*)d_in[5];
  const float* Wk = (const float*)d_in[6];
  const float* bk = (const float*)d_in[7];
  const float* Wv = (const float*)d_in[8];
  const float* bv = (const float*)d_in[9];
  const float* Wo = (const float*)d_in[10];
  const float* bo = (const float*)d_in[11];
  float*  out = (float*)d_out;
  __bf16* ws  = (__bf16*)d_ws;

  // 1) fp32 -> bf16 staging
  cvt_kernel<<<dim3(2048), dim3(256), 0, stream>>>(x, Wq, Wk, Wv, Wo, ws);
  // 2) QKV projection: 3 * 512 * 12 = 18432 waves, 4 waves/block
  qkv_kernel<<<dim3(4608), dim3(128), 0, stream>>>(bq, bk, bv, ws);
  // 3) flash attention: 8*12*64 = 6144 waves
  attn_kernel<<<dim3(1536), dim3(128), 0, stream>>>(sp, ee, mask, ws);
  // 4) output projection: 512 * 12 = 6144 waves
  out_kernel<<<dim3(1536), dim3(128), 0, stream>>>(bo, ws, out);
}
